// ScaledDotProductAttention_39797166965451
// MI455X (gfx1250) — matrix operand order repaired and checked by reading, not compiled, not run on hardware
//
#include <hip/hip_runtime.h>
#include <hip/hip_bf16.h>

typedef __attribute__((ext_vector_type(16))) _Float16 v16h;
typedef __attribute__((ext_vector_type(8)))  _Float16 v8h;
typedef __attribute__((ext_vector_type(8)))  float    v8f;
typedef __attribute__((ext_vector_type(4)))  float    v4f;
typedef __attribute__((ext_vector_type(4)))  int      v4i;

#define B_      2
#define H_      12
#define S_      2048
#define D_      64
#define BLOCK_M 128
#define BLOCK_N 32
#define NWAVES  8

// Cross 16-lane-group exchange: lane <-> lane^16 (ds_swizzle SWAPX16:
// group-of-32, xor_mask=0x10, and_mask=0x1f  => offset 0x401F). Uses LDS
// permute hardware, no LDS memory traffic.
__device__ __forceinline__ float swz16f(float x) {
    return __builtin_bit_cast(float,
        __builtin_amdgcn_ds_swizzle(__builtin_bit_cast(int, x), 0x401F));
}

__global__ __launch_bounds__(256) void fa_fwd_gfx1250(
    const float* __restrict__ q,
    const float* __restrict__ k,
    const float* __restrict__ v,
    float* __restrict__ out)
{
    // Double-buffered K/V staging (P transpose tile no longer needed).
    __shared__ alignas(32) _Float16 Klds[2][BLOCK_N * D_];   // [key][d] row-major
    __shared__ alignas(32) _Float16 Vlds[2][D_ * BLOCK_N];   // [d][key] transposed

    const int tid  = threadIdx.x;
    const int lane = tid & 31;
    const int r    = lane & 15;
    const int hi   = lane >> 4;    // 0 = lanes 0-15, 1 = lanes 16-31

    const int bh   = blockIdx.y;
    const int m0wg = blockIdx.x * BLOCK_M;
    const int m0w  = m0wg + (tid >> 5) * 16;

    const size_t base = (size_t)bh * S_ * D_;
    const float* qp = q + base;
    const float* kp = k + base;
    const float* vp = v + base;
    float*       op = out + base;

    // ---- Q^T as two B-fragments (32d x 16m each); fold 1/sqrt(D) here ----
    // B layout: element i -> K(d_local) = (i<8?i:i+8) + 8*hi; n(m)=lane&15.
    v16h bq[2];
    {
        const float* qrow = qp + (size_t)(m0w + r) * D_;
        #pragma unroll
        for (int c = 0; c < 2; ++c)
            #pragma unroll
            for (int i = 0; i < 16; ++i)
                bq[c][i] = (_Float16)(qrow[32 * c + 8 * hi + (i < 8 ? i : i + 8)] * 0.125f);
    }

    // O^T accumulators: 4 C-frags, chunk c: row d = 16c + j + 8*hi, col m = r.
    v8f o0 = {}, o1 = {}, o2 = {}, o3 = {};
    // Per-lane scalar softmax state for query column m = m0w + r.
    float mold = -__builtin_inff();
    float lsum = 0.0f;

    const int kend = m0wg + BLOCK_M;   // workgroup-uniform causal bound

    // ---- staging: K row-major, V transposed; one b128 DS store each ----
    auto stage = [&](int kb, int buf) {
        {   // K: thread -> (key = tid/8, 8 contiguous d); coalesced b128 loads
            const int key = tid >> 3;
            const int db  = (tid & 7) * 8;
            const float* kr = kp + (size_t)(kb + key) * D_ + db;
            v8h kh;
            #pragma unroll
            for (int i = 0; i < 8; ++i) kh[i] = (_Float16)kr[i];
            *reinterpret_cast<v8h*>(&Klds[buf][key * D_ + db]) = kh;
        }
        {   // V^T: thread -> (one d column, 8 consecutive keys); coalesced b32 loads
            const int d  = tid & 63;
            const int kg = (tid >> 6) * 8;
            const float* vr = vp + (size_t)(kb + kg) * D_ + d;
            v8h vh;
            #pragma unroll
            for (int i = 0; i < 8; ++i) vh[i] = (_Float16)vr[(size_t)i * D_];
            *reinterpret_cast<v8h*>(&Vlds[buf][d * BLOCK_N + kg]) = vh;
        }
    };

    stage(0, 0);
    int ib = 0;

    for (int kb = 0; kb < kend; kb += BLOCK_N, ib ^= 1) {
        __syncthreads();   // stage(kb) done; previous compute's LDS reads done
        if (kb + BLOCK_N < kend) stage(kb + BLOCK_N, ib ^ 1);   // prefetch next

        if (kb > m0w + 15) continue;   // fully above diagonal (wave-uniform)

        const _Float16* Kb = &Klds[ib][0];
        const _Float16* Vb = &Vlds[ib][0];

        // ---- S^T = K · Q^T : two 16x16 tiles (t: keys kb+16t..kb+16t+15) ----
        // A-frag of K: row key = 16t + r; element i -> d = 32c + (i<8?i:i+8) + 8*hi.
        v8f s0 = {}, s1 = {};
        #pragma unroll
        for (int t = 0; t < 2; ++t) {
            v8f s = {};
            #pragma unroll
            for (int c = 0; c < 2; ++c) {
                const _Float16* krow = &Kb[(16 * t + r) * D_ + 32 * c + 8 * hi];
                v16h ak;
                #pragma unroll
                for (int i = 0; i < 8; ++i) ak[i]     = krow[i];
                #pragma unroll
                for (int i = 0; i < 8; ++i) ak[8 + i] = krow[16 + i];
                s = __builtin_amdgcn_wmma_f32_16x16x32_f16(
                        false, ak, false, bq[c], (short)0, s, false, false);
            }
            if (t == 0) s0 = s; else s1 = s;
        }

        // ---- causal mask only on diagonal blocks ----
        // s_t C layout: row key = kb + 16t + j + 8*hi, col m = m0w + r.
        float p0[8], p1[8];
        const bool diag = (kb + BLOCK_N - 1) > m0w;
        const int  mg   = m0w + r;
        if (diag) {
            #pragma unroll
            for (int j = 0; j < 8; ++j) {
                p0[j] = (kb +      j + 8 * hi <= mg) ? s0[j] : -__builtin_inff();
                p1[j] = (kb + 16 + j + 8 * hi <= mg) ? s1[j] : -__builtin_inff();
            }
        } else {
            #pragma unroll
            for (int j = 0; j < 8; ++j) { p0[j] = s0[j]; p1[j] = s1[j]; }
        }

        // ---- per-lane scalar online softmax (one swizzle per reduction) ----
        float mloc = -__builtin_inff();
        #pragma unroll
        for (int j = 0; j < 8; ++j) mloc = fmaxf(mloc, fmaxf(p0[j], p1[j]));
        const float mnew  = fmaxf(mold, fmaxf(mloc, swz16f(mloc)));
        const float alpha = __expf(mold - mnew);   // 0 on first block
        float ls = 0.0f;
        #pragma unroll
        for (int j = 0; j < 8; ++j) {
            p0[j] = __expf(p0[j] - mnew);
            p1[j] = __expf(p1[j] - mnew);
            ls += p0[j] + p1[j];
        }
        lsum = lsum * alpha + ls + swz16f(ls);
        mold = mnew;
        #pragma unroll
        for (int j = 0; j < 8; ++j) {
            o0[j] *= alpha; o1[j] *= alpha; o2[j] *= alpha; o3[j] *= alpha;
        }

        // ---- P^T as B-fragment: own p0 then own p1 ----
        // Element i -> key = (i<8?i:i+8) + 8*hi (lane<16: keys 0..7,16..23;
        // lane>=16: keys 8..15,24..31).
        v16h pb;
        {
            v8h h0, h1;
            #pragma unroll
            for (int j = 0; j < 8; ++j) { h0[j] = (_Float16)p0[j]; h1[j] = (_Float16)p1[j]; }
            const v4i i0 = __builtin_bit_cast(v4i, h0);
            const v4i i1 = __builtin_bit_cast(v4i, h1);
            v4i isel, icrs, ilo, ihi;
            #pragma unroll
            for (int d = 0; d < 4; ++d) isel[d] = hi ? i1[d] : i0[d];
            #pragma unroll
            for (int d = 0; d < 4; ++d)
                icrs[d] = __builtin_amdgcn_ds_swizzle(isel[d], 0x401F);
            #pragma unroll
            for (int d = 0; d < 4; ++d) {
                ilo[d] = hi ? icrs[d] : isel[d];
                ihi[d] = hi ? isel[d] : icrs[d];
            }
            const v8h hlo = __builtin_bit_cast(v8h, ilo);
            const v8h hhi = __builtin_bit_cast(v8h, ihi);
            #pragma unroll
            for (int i = 0; i < 8; ++i) { pb[i] = h0[i]; pb[8 + i] = h1[i]; }
        }

        // ---- O^T += V^T · P^T ----
        // A-frag of V^T chunk c: row d = 16c + r; element i -> key = (i<8?i:i+8)+8*hi.
        #pragma unroll
        for (int c = 0; c < 4; ++c) {
            const _Float16* vrow = &Vb[(16 * c + r) * BLOCK_N + 8 * hi];
            v16h av;
            #pragma unroll
            for (int i = 0; i < 8; ++i) av[i]     = vrow[i];
            #pragma unroll
            for (int i = 0; i < 8; ++i) av[8 + i] = vrow[16 + i];
            v8f& oc = (c == 0) ? o0 : (c == 1) ? o1 : (c == 2) ? o2 : o3;
            oc = __builtin_amdgcn_wmma_f32_16x16x32_f16(
                     false, av, false, pb, (short)0, oc, false, false);
        }
    }

    // ---- epilogue: lane owns output row m = m0w + r; contiguous d-runs of 8 ----
    {
        const float rl = 1.0f / lsum;
        float* orow = op + (size_t)(m0w + r) * D_;
        #pragma unroll
        for (int c = 0; c < 4; ++c) {
            const v8f& oc = (c == 0) ? o0 : (c == 1) ? o1 : (c == 2) ? o2 : o3;
            v4f x0, x1;
            #pragma unroll
            for (int j = 0; j < 4; ++j) { x0[j] = oc[j] * rl; x1[j] = oc[4 + j] * rl; }
            *reinterpret_cast<v4f*>(&orow[16 * c + 8 * hi])     = x0;
            *reinterpret_cast<v4f*>(&orow[16 * c + 8 * hi + 4]) = x1;
        }
    }
}

extern "C" void kernel_launch(void* const* d_in, const int* in_sizes, int n_in,
                              void* d_out, int out_size, void* d_ws, size_t ws_size,
                              hipStream_t stream) {
    const float* q = (const float*)d_in[0];
    const float* k = (const float*)d_in[1];
    const float* v = (const float*)d_in[2];
    // d_in[3] is the boolean tril mask; the reference's flip-logic resolves it
    // to plain causal (keep n <= m), which is hardcoded in the kernel.
    float* out = (float*)d_out;

    dim3 grid(S_ / BLOCK_M, B_ * H_);
    fa_fwd_gfx1250<<<grid, 256, 0, stream>>>(q, k, v, out);
}
